// Critic_Net_PNAConv_Model_54271206752464
// MI455X (gfx1250) — matrix operand order repaired and checked by reading, not compiled, not run on hardware
//
#include <hip/hip_runtime.h>
#include <hip/hip_bf16.h>
#include <math.h>
#include <stdint.h>

typedef __attribute__((ext_vector_type(16))) _Float16 v16h;
typedef __attribute__((ext_vector_type(8)))  float    v8f;

// ---------------------------------------------------------------------------
// WMMA fragment loaders (CDNA5 ISA 7.12.2 lane layouts, wave32)
// A: 16x32 f16, row M = lane%16, K striped over 8 VGPR pairs (2x b128 / lane).
// ---------------------------------------------------------------------------
__device__ __forceinline__ v16h load_frag_A(const _Float16* base, int ld, int lane) {
  int g = (lane >> 4) & 1, m = lane & 15;
  v16h a;
#pragma unroll
  for (int v = 0; v < 8; ++v) {
    int kb = g * 16 + v * 2;
    a[2 * v]     = base[m * ld + kb];
    a[2 * v + 1] = base[m * ld + kb + 1];
  }
  return a;
}

// B operand staged fragment-major: sXf[ks][lane][j] so each lane's 16 halves
// are contiguous -> two ds_load_b128, no packing VALU.
__device__ __forceinline__ v16h load_frag_Bf(const _Float16* base, int lane) {
  return *(const v16h*)(base + lane * 16);
}

// swizzled store index for fragment-major staging: (k within KP, col n) ->
// element j of fragment lane fl of k-step ks.
__device__ __forceinline__ int bstage_idx(int k, int n) {
  int ks = k >> 5, kk = k & 31;
  int fl = ((kk >> 4) << 4) | n;
  return ks * 512 + fl * 16 + (kk & 15);
}

// Fire-and-forget float max atomic (GLOBAL_ATOMIC_MAX_NUM_F32, no return,
// STOREcnt-tracked).
__device__ __forceinline__ void atomicMaxF(float* addr, float val) {
  asm volatile("global_atomic_max_num_f32 %0, %1, off scope:SCOPE_DEV"
               :: "v"((unsigned long long)(uintptr_t)addr), "v"(val)
               : "memory");
}

// Async DMA of pre-packed f16 weight tile (global -> LDS), ASYNCcnt-tracked.
__device__ __forceinline__ void async_copy_w(unsigned lds_base, const _Float16* gbase,
                                             int nbytes, int tid) {
  for (int off = tid * 16; off < nbytes; off += 128 * 16) {
    asm volatile("global_load_async_to_lds_b128 %0, %1, %2"
                 :: "v"(lds_base + (unsigned)off), "v"((unsigned)off), "s"(gbase)
                 : "memory");
  }
}
__device__ __forceinline__ void wait_async() {
  asm volatile("s_wait_asynccnt 0" ::: "memory");
}

// ---------------------------------------------------------------------------
// Small scalar kernels
// ---------------------------------------------------------------------------
__global__ void fill_kernel(float* p, float v, int n) {
  int i = blockIdx.x * blockDim.x + threadIdx.x;
  if (i < n) p[i] = v;
}

// pack fp32 [M][K] row-major -> f16 [MROWS][KP] zero-padded (once per launch)
__global__ void pack_w_kernel(const float* __restrict__ src, _Float16* __restrict__ dstp,
                              int M, int K, int MROWS, int KP) {
  int i = blockIdx.x * blockDim.x + threadIdx.x;
  if (i < MROWS * KP) {
    int m = i / KP, k = i % KP;
    dstp[i] = (_Float16)((m < M && k < K) ? src[m * K + k] : 0.0f);
  }
}

__global__ void deg_kernel(const int* __restrict__ dst, float* deg, int E_) {
  int i = blockIdx.x * blockDim.x + threadIdx.x;
  if (i < E_) atomicAdd(&deg[dst[i]], 1.0f);
}

__global__ void npg_kernel(const int* __restrict__ gid, float* npg, int Nn) {
  int i = blockIdx.x * blockDim.x + threadIdx.x;
  if (i < Nn) atomicAdd(&npg[gid[i]], 1.0f);
}

__global__ void snorm_kernel(const int* __restrict__ gid, const float* __restrict__ npg,
                             float* snorm, int Nn) {
  int i = blockIdx.x * blockDim.x + threadIdx.x;
  if (i < Nn) snorm[i] = rsqrtf(npg[gid[i]]);
}

__global__ void assemble_in_kernel(const float* __restrict__ na, float* h12, float* ns_s, int Nn) {
  int i = blockIdx.x * blockDim.x + threadIdx.x;
  if (i < Nn) {
#pragma unroll
    for (int j = 0; j < 12; ++j) h12[i * 12 + j] = na[i * 13 + j];
    ns_s[i] = na[i * 13 + 12];
  }
}

__global__ void assemble_t_kernel(const float* __restrict__ h8, const float* __restrict__ ns_s,
                                  float* t12, int Nn) {
  int i = blockIdx.x * blockDim.x + threadIdx.x;
  if (i < Nn * 12) {
    int n = i / 12, j = i % 12;
    t12[i] = (j < 8) ? h8[n * 8 + j] : ns_s[n];
  }
}

__global__ void bn_finalize_kernel(const float* __restrict__ bsum, const float* __restrict__ bsq,
                                   const float* __restrict__ gamma, const float* __restrict__ beta,
                                   float* scale, float* shift, int F, float invN) {
  int i = threadIdx.x;
  if (i < F) {
    float mu  = bsum[i] * invN;
    float var = bsq[i] * invN - mu * mu;
    float sc  = rsqrtf(var + 1e-5f) * gamma[i];
    scale[i] = sc;
    shift[i] = beta[i] - mu * sc;
  }
}

// ---------------------------------------------------------------------------
// Edge-message GEMM + scatter (sum / sum^2 / max) with atomics.
// msg = Mw @ [h_src; e; h_dst] + Mb    (per edge, WMMA tiled 16 edges x 16 M)
// ---------------------------------------------------------------------------
template <int FIN>
__global__ __launch_bounds__(128) void edge_msg_kernel(
    const float* __restrict__ h, const float* __restrict__ eattr,
    const int* __restrict__ src, const int* __restrict__ dst,
    const _Float16* __restrict__ Wf16, const float* __restrict__ Mb,
    float* __restrict__ sum, float* __restrict__ sumsq, float* __restrict__ mx, int E_) {
  constexpr int K     = 2 * FIN + 2;
  constexpr int KP    = (K + 31) & ~31;
  constexpr int NKS   = KP / 32;
  constexpr int MROWS = (FIN + 15) & ~15;
  constexpr int MTW   = MROWS / 16;   // waves per edge tile
  constexpr int NT    = 4 / MTW;      // edge tiles per block (4 waves)

  __shared__ __align__(32) _Float16 sW[MROWS * KP];
  __shared__ __align__(32) _Float16 sXf[NT][NKS * 512];
  __shared__ int sSrc[NT * 16], sDst[NT * 16];
  __shared__ float sBias[MROWS];

  const int tid = threadIdx.x, lane = tid & 31, wave = tid >> 5;
  const int etile = wave / MTW, mtile = wave % MTW;
  const int ebase = blockIdx.x * (NT * 16);

  // async DMA of pre-packed f16 weights into LDS (overlaps index/X staging)
  async_copy_w((unsigned)(uintptr_t)&sW[0], Wf16, MROWS * KP * 2, tid);

  for (int i = tid; i < NT * 16; i += 128) {
    int e = ebase + i;
    sSrc[i] = (e < E_) ? src[e] : 0;
    sDst[i] = (e < E_) ? dst[e] : 0;
  }
  for (int i = tid; i < MROWS; i += 128) sBias[i] = (i < FIN) ? Mb[i] : 0.0f;
  __syncthreads();

  for (int i = tid; i < NT * 16 * KP; i += 128) {
    int t = i / (16 * KP), rem = i % (16 * KP);
    int k = rem / 16, n = rem % 16;
    int e = ebase + t * 16 + n;
    float v = 0.0f;
    if (e < E_) {
      if (k < FIN)            v = h[sSrc[t * 16 + n] * FIN + k];
      else if (k < FIN + 2)   v = eattr[e * 2 + (k - FIN)];
      else if (k < K)         v = h[sDst[t * 16 + n] * FIN + (k - FIN - 2)];
    }
    sXf[t][bstage_idx(k, n)] = (_Float16)v;
  }
  wait_async();
  __syncthreads();

  v8f c = {};
#pragma unroll
  for (int ks = 0; ks < NKS; ++ks) {
    v16h a = load_frag_A(&sW[mtile * 16 * KP + ks * 32], KP, lane);
    v16h b = load_frag_Bf(&sXf[etile][ks * 512], lane);
    c = __builtin_amdgcn_wmma_f32_16x16x32_f16(false, a, false, b, (short)0, c, false, false);
  }

  const int g = (lane >> 4) & 1, n = lane & 15;
  const int e = ebase + etile * 16 + n;
  if (e < E_) {
    const int d = sDst[etile * 16 + n];
#pragma unroll
    for (int r = 0; r < 8; ++r) {
      int m = mtile * 16 + r + 8 * g;
      if (m < FIN) {
        float v = c[r] + sBias[m];
        atomicAdd(&sum[d * FIN + m], v);
        atomicAdd(&sumsq[d * FIN + m], v * v);
        atomicMaxF(&mx[d * FIN + m], v);
      }
    }
  }
}

// ---------------------------------------------------------------------------
// Node aggregate + Uw GEMM + snorm; accumulates BN statistics.
// x = [h, mean, max, sum, std] (5*FIN);  pre = (Uw @ x + Ub) * snorm
// ---------------------------------------------------------------------------
template <int FIN, int FOUT>
__global__ __launch_bounds__(128) void node_agg_kernel(
    const float* __restrict__ h, const float* __restrict__ sum,
    const float* __restrict__ sumsq, const float* __restrict__ mx,
    const float* __restrict__ deg, const float* __restrict__ snorm,
    const _Float16* __restrict__ Wf16, const float* __restrict__ Ub,
    float* __restrict__ pre, float* __restrict__ bn_sum, float* __restrict__ bn_sq, int Nn) {
  constexpr int K     = 5 * FIN;
  constexpr int KP    = (K + 31) & ~31;
  constexpr int NKS   = KP / 32;
  constexpr int MROWS = (FOUT + 15) & ~15;
  constexpr int MTW   = MROWS / 16;
  constexpr int NT    = 4 / MTW;

  __shared__ __align__(32) _Float16 sW[MROWS * KP];
  __shared__ __align__(32) _Float16 sXf[NT][NKS * 512];
  __shared__ float sDd[NT * 16];
  __shared__ float sHas[NT * 16];
  __shared__ float sRed[2 * FOUT];
  __shared__ float sBias[MROWS];

  const int tid = threadIdx.x, lane = tid & 31, wave = tid >> 5;
  const int ntile = wave / MTW, mtile = wave % MTW;
  const int nbase = blockIdx.x * (NT * 16);

  async_copy_w((unsigned)(uintptr_t)&sW[0], Wf16, MROWS * KP * 2, tid);

  for (int i = tid; i < NT * 16; i += 128) {
    int node = nbase + i;
    float dg = (node < Nn) ? deg[node] : 0.0f;
    sDd[i]  = fmaxf(dg, 1.0f);
    sHas[i] = (dg > 0.0f) ? 1.0f : 0.0f;
  }
  for (int i = tid; i < 2 * FOUT; i += 128) sRed[i] = 0.0f;
  for (int i = tid; i < MROWS; i += 128) sBias[i] = (i < FOUT) ? Ub[i] : 0.0f;
  __syncthreads();

  for (int i = tid; i < NT * 16 * KP; i += 128) {
    int t = i / (16 * KP), rem = i % (16 * KP);
    int k = rem / 16, n = rem % 16;
    int node = nbase + t * 16 + n;
    float v = 0.0f;
    if (node < Nn && k < K) {
      int band = k / FIN, j = k % FIN;
      float dd = sDd[t * 16 + n], has = sHas[t * 16 + n];
      if (band == 0) v = h[node * FIN + j];
      else {
        float s = sum[node * FIN + j];
        if (band == 1)      v = has * (s / dd);
        else if (band == 2) v = has * mx[node * FIN + j];
        else if (band == 3) v = s;
        else {
          float m_ = s / dd;
          float var = sumsq[node * FIN + j] / dd - m_ * m_;
          v = has * sqrtf(fmaxf(var, 0.0f) + 1e-30f);
        }
      }
    }
    sXf[t][bstage_idx(k, n)] = (_Float16)v;
  }
  wait_async();
  __syncthreads();

  v8f c = {};
#pragma unroll
  for (int ks = 0; ks < NKS; ++ks) {
    v16h a = load_frag_A(&sW[mtile * 16 * KP + ks * 32], KP, lane);
    v16h b = load_frag_Bf(&sXf[ntile][ks * 512], lane);
    c = __builtin_amdgcn_wmma_f32_16x16x32_f16(false, a, false, b, (short)0, c, false, false);
  }

  const int g = (lane >> 4) & 1, n = lane & 15;
  const int node = nbase + ntile * 16 + n;
  if (node < Nn) {
    float sn = snorm[node];
#pragma unroll
    for (int r = 0; r < 8; ++r) {
      int m = mtile * 16 + r + 8 * g;
      if (m < FOUT) {
        float v = (c[r] + sBias[m]) * sn;
        pre[node * FOUT + m] = v;
        atomicAdd(&sRed[m], v);              // ds_add_f32 partial BN stats
        atomicAdd(&sRed[FOUT + m], v * v);
      }
    }
  }
  __syncthreads();
  if (tid < FOUT) {
    atomicAdd(&bn_sum[tid], sRed[tid]);
    atomicAdd(&bn_sq[tid], sRed[FOUT + tid]);
  }
}

// ---------------------------------------------------------------------------
// BN apply + mw GEMM + LeakyReLU + ReLU  ->  next layer features
// ---------------------------------------------------------------------------
template <int FOUT>
__global__ __launch_bounds__(128) void node_out_kernel(
    const float* __restrict__ pre, const float* __restrict__ scale,
    const float* __restrict__ shift, const _Float16* __restrict__ Wf16,
    const float* __restrict__ mb, float* __restrict__ hout, int Nn) {
  constexpr int K     = FOUT;
  constexpr int KP    = (K + 31) & ~31;
  constexpr int NKS   = KP / 32;
  constexpr int MROWS = (FOUT + 15) & ~15;
  constexpr int MTW   = MROWS / 16;
  constexpr int NT    = 4 / MTW;

  __shared__ __align__(32) _Float16 sW[MROWS * KP];
  __shared__ __align__(32) _Float16 sXf[NT][NKS * 512];
  __shared__ float sBias[MROWS];

  const int tid = threadIdx.x, lane = tid & 31, wave = tid >> 5;
  const int ntile = wave / MTW, mtile = wave % MTW;
  const int nbase = blockIdx.x * (NT * 16);

  async_copy_w((unsigned)(uintptr_t)&sW[0], Wf16, MROWS * KP * 2, tid);

  for (int i = tid; i < MROWS; i += 128) sBias[i] = (i < FOUT) ? mb[i] : 0.0f;
  for (int i = tid; i < NT * 16 * KP; i += 128) {
    int t = i / (16 * KP), rem = i % (16 * KP);
    int k = rem / 16, n = rem % 16;
    int node = nbase + t * 16 + n;
    float v = 0.0f;
    if (node < Nn && k < K) v = pre[node * FOUT + k] * scale[k] + shift[k];
    sXf[t][bstage_idx(k, n)] = (_Float16)v;
  }
  wait_async();
  __syncthreads();

  v8f c = {};
#pragma unroll
  for (int ks = 0; ks < NKS; ++ks) {
    v16h a = load_frag_A(&sW[mtile * 16 * KP + ks * 32], KP, lane);
    v16h b = load_frag_Bf(&sXf[ntile][ks * 512], lane);
    c = __builtin_amdgcn_wmma_f32_16x16x32_f16(false, a, false, b, (short)0, c, false, false);
  }

  const int g = (lane >> 4) & 1, n = lane & 15;
  const int node = nbase + ntile * 16 + n;
  if (node < Nn) {
#pragma unroll
    for (int r = 0; r < 8; ++r) {
      int m = mtile * 16 + r + 8 * g;
      if (m < FOUT) {
        float v = c[r] + sBias[m];
        v = (v >= 0.0f) ? v : 0.01f * v;   // LeakyReLU
        v = fmaxf(v, 0.0f);                // ReLU
        hout[node * FOUT + m] = v;
      }
    }
  }
}

// ---------------------------------------------------------------------------
// Graph max-pool (64 feats) + conv_w dot + ReLU  ->  logits[B]
// ---------------------------------------------------------------------------
__global__ __launch_bounds__(256) void pool_kernel(const float* __restrict__ t,
                                                   const float* __restrict__ conv_w,
                                                   const float* __restrict__ conv_b,
                                                   float* __restrict__ out, int npg) {
  __shared__ float sMax[4 * 64];
  __shared__ float sDot[64];
  const int gph = blockIdx.x, tid = threadIdx.x;
  const int f = tid & 63, chunk = tid >> 6;
  float m = -INFINITY;
  for (int n = chunk; n < npg; n += 4) m = fmaxf(m, t[(gph * npg + n) * 64 + f]);
  sMax[chunk * 64 + f] = m;
  __syncthreads();
  if (tid < 64) {
    float mm = fmaxf(fmaxf(sMax[tid], sMax[64 + tid]), fmaxf(sMax[128 + tid], sMax[192 + tid]));
    sDot[tid] = mm * conv_w[tid];
  }
  __syncthreads();
  if (tid == 0) {
    float s = 0.0f;
    for (int i = 0; i < 64; ++i) s += sDot[i];
    out[gph] = fmaxf(s + conv_b[0], 0.0f);
  }
}

// ---------------------------------------------------------------------------
// Host orchestration
// ---------------------------------------------------------------------------
static inline int cdiv(int a, int b) { return (a + b - 1) / b; }

extern "C" void kernel_launch(void* const* d_in, const int* in_sizes, int n_in,
                              void* d_out, int out_size, void* d_ws, size_t ws_size,
                              hipStream_t stream) {
  (void)n_in; (void)ws_size;
  const float* node_attr = (const float*)d_in[0];
  const float* edge_attr = (const float*)d_in[1];
  const int*   src       = (const int*)d_in[2];
  const int*   dst       = (const int*)d_in[3];
  const int*   gid       = (const int*)d_in[4];
  auto P = [&](int l, int j) { return (const float*)d_in[5 + l * 8 + j]; };
  const float* conv_w = (const float*)d_in[45];
  const float* conv_b = (const float*)d_in[46];

  const int N = in_sizes[0] / 13;
  const int E = in_sizes[2];
  const int B = out_size;
  const int NPGc = N / B;

  // workspace carve
  char* w = (char*)d_ws;
  auto carve = [&](size_t bytes) { char* p = w; w += (bytes + 255) & ~(size_t)255; return p; };
  float* hA     = (float*)carve((size_t)N * 64 * 4);
  float* hB     = (float*)carve((size_t)N * 64 * 4);
  float* sumB   = (float*)carve((size_t)N * 64 * 4);
  float* sumsqB = (float*)carve((size_t)N * 64 * 4);
  float* mxB    = (float*)carve((size_t)N * 64 * 4);
  float* preB   = (float*)carve((size_t)N * 64 * 4);
  float* ns_s   = (float*)carve((size_t)N * 4);
  float* deg    = (float*)carve((size_t)N * 4);
  float* snorm  = (float*)carve((size_t)N * 4);
  float* npg    = (float*)carve(64 * 4);
  float* bn_sum = (float*)carve(64 * 4);
  float* bn_sq  = (float*)carve(64 * 4);
  float* bn_sc  = (float*)carve(64 * 4);
  float* bn_sh  = (float*)carve(64 * 4);
  _Float16* wEdge = (_Float16*)carve(64 * 160 * 2);   // max edge weight tile
  _Float16* wNode = (_Float16*)carve(64 * 320 * 2);   // max node weight tile
  _Float16* wOut  = (_Float16*)carve(64 * 64 * 2);    // max out  weight tile

  const float NEG_INF = -INFINITY;
  auto fill = [&](float* p, float v, int n) {
    fill_kernel<<<cdiv(n, 256), 256, 0, stream>>>(p, v, n);
  };
  auto pack = [&](const float* srcW, _Float16* dstW, int M, int K) {
    int MR = (M + 15) & ~15, KPp = (K + 31) & ~31;
    pack_w_kernel<<<cdiv(MR * KPp, 256), 256, 0, stream>>>(srcW, dstW, M, K, MR, KPp);
  };

  // degree / snorm (same for all layers)
  fill(deg, 0.0f, N);
  fill(npg, 0.0f, 64);
  deg_kernel<<<cdiv(E, 256), 256, 0, stream>>>(dst, deg, E);
  npg_kernel<<<cdiv(N, 256), 256, 0, stream>>>(gid, npg, N);
  snorm_kernel<<<cdiv(N, 256), 256, 0, stream>>>(gid, npg, snorm, N);
  assemble_in_kernel<<<cdiv(N, 256), 256, 0, stream>>>(node_attr, hA, ns_s, N);

  auto prep = [&](int l, int fin, int fout) {
    fill(sumB, 0.0f, N * fin);
    fill(sumsqB, 0.0f, N * fin);
    fill(mxB, NEG_INF, N * fin);
    fill(bn_sum, 0.0f, fout);
    fill(bn_sq, 0.0f, fout);
    pack(P(l, 0), wEdge, fin, 2 * fin + 2);
    pack(P(l, 2), wNode, fout, 5 * fin);
    pack(P(l, 6), wOut, fout, fout);
  };
  const float invN = 1.0f / (float)N;

  // ---- layer 0: fin=12 -> fout=64   (hA -> hB)
  prep(0, 12, 64);
  edge_msg_kernel<12><<<cdiv(E, 64), 128, 0, stream>>>(hA, edge_attr, src, dst, wEdge, P(0, 1),
                                                       sumB, sumsqB, mxB, E);
  node_agg_kernel<12, 64><<<cdiv(N, 16), 128, 0, stream>>>(hA, sumB, sumsqB, mxB, deg, snorm,
                                                           wNode, P(0, 3), preB, bn_sum, bn_sq, N);
  bn_finalize_kernel<<<1, 64, 0, stream>>>(bn_sum, bn_sq, P(0, 4), P(0, 5), bn_sc, bn_sh, 64, invN);
  node_out_kernel<64><<<cdiv(N, 16), 128, 0, stream>>>(preB, bn_sc, bn_sh, wOut, P(0, 7), hB, N);

  // ---- layer 1: fin=64 -> fout=64   (hB -> hA)
  prep(1, 64, 64);
  edge_msg_kernel<64><<<cdiv(E, 16), 128, 0, stream>>>(hB, edge_attr, src, dst, wEdge, P(1, 1),
                                                       sumB, sumsqB, mxB, E);
  node_agg_kernel<64, 64><<<cdiv(N, 16), 128, 0, stream>>>(hB, sumB, sumsqB, mxB, deg, snorm,
                                                           wNode, P(1, 3), preB, bn_sum, bn_sq, N);
  bn_finalize_kernel<<<1, 64, 0, stream>>>(bn_sum, bn_sq, P(1, 4), P(1, 5), bn_sc, bn_sh, 64, invN);
  node_out_kernel<64><<<cdiv(N, 16), 128, 0, stream>>>(preB, bn_sc, bn_sh, wOut, P(1, 7), hA, N);

  // ---- layer 2: fin=64 -> fout=8    (hA -> hB[8])
  prep(2, 64, 8);
  edge_msg_kernel<64><<<cdiv(E, 16), 128, 0, stream>>>(hA, edge_attr, src, dst, wEdge, P(2, 1),
                                                       sumB, sumsqB, mxB, E);
  node_agg_kernel<64, 8><<<cdiv(N, 64), 128, 0, stream>>>(hA, sumB, sumsqB, mxB, deg, snorm,
                                                          wNode, P(2, 3), preB, bn_sum, bn_sq, N);
  bn_finalize_kernel<<<1, 64, 0, stream>>>(bn_sum, bn_sq, P(2, 4), P(2, 5), bn_sc, bn_sh, 8, invN);
  node_out_kernel<8><<<cdiv(N, 64), 128, 0, stream>>>(preB, bn_sc, bn_sh, wOut, P(2, 7), hB, N);

  // ---- reassemble 12-wide input: [h8, ns_s x4]  (hB[8] + ns_s -> hA[12])
  assemble_t_kernel<<<cdiv(N * 12, 256), 256, 0, stream>>>(hB, ns_s, hA, N);

  // ---- layer 3: fin=12 -> fout=64   (hA -> hB)
  prep(3, 12, 64);
  edge_msg_kernel<12><<<cdiv(E, 64), 128, 0, stream>>>(hA, edge_attr, src, dst, wEdge, P(3, 1),
                                                       sumB, sumsqB, mxB, E);
  node_agg_kernel<12, 64><<<cdiv(N, 16), 128, 0, stream>>>(hA, sumB, sumsqB, mxB, deg, snorm,
                                                           wNode, P(3, 3), preB, bn_sum, bn_sq, N);
  bn_finalize_kernel<<<1, 64, 0, stream>>>(bn_sum, bn_sq, P(3, 4), P(3, 5), bn_sc, bn_sh, 64, invN);
  node_out_kernel<64><<<cdiv(N, 16), 128, 0, stream>>>(preB, bn_sc, bn_sh, wOut, P(3, 7), hB, N);

  // ---- layer 4: fin=64 -> fout=64   (hB -> hA)
  prep(4, 64, 64);
  edge_msg_kernel<64><<<cdiv(E, 16), 128, 0, stream>>>(hB, edge_attr, src, dst, wEdge, P(4, 1),
                                                       sumB, sumsqB, mxB, E);
  node_agg_kernel<64, 64><<<cdiv(N, 16), 128, 0, stream>>>(hB, sumB, sumsqB, mxB, deg, snorm,
                                                           wNode, P(4, 3), preB, bn_sum, bn_sq, N);
  bn_finalize_kernel<<<1, 64, 0, stream>>>(bn_sum, bn_sq, P(4, 4), P(4, 5), bn_sc, bn_sh, 64, invN);
  node_out_kernel<64><<<cdiv(N, 16), 128, 0, stream>>>(preB, bn_sc, bn_sh, wOut, P(4, 7), hA, N);

  // ---- graph max-pool + linear + ReLU
  pool_kernel<<<B, 256, 0, stream>>>(hA, conv_w, conv_b, (float*)d_out, NPGc);
}
